// Model_55594056680038
// MI455X (gfx1250) — compile-verified, matrix operand order checked by reading, not run on hardware
//
#include <hip/hip_runtime.h>

typedef __attribute__((ext_vector_type(16))) __bf16 v16bf;
typedef __attribute__((ext_vector_type(8)))  float  v8f;

#define NNODES 50000
#define NEDGES 800000
#define DIN    64
#define EDIMC  64
#define HIDC   128
#define DOUTC  128
#define NCLS   10

__device__ __forceinline__ void atomic_add_f32(float* p, float v) {
    __hip_atomic_fetch_add(p, v, __ATOMIC_RELAXED, __HIP_MEMORY_SCOPE_AGENT);
}

// pack 8 f32 -> 8 bf16 (16 bytes); compiler emits 4x v_cvt_pk_bf16_f32
__device__ __forceinline__ uint4 cvt8(float4 a, float4 b) {
    union { __bf16 h[8]; uint4 u; } r;
    r.h[0] = (__bf16)a.x; r.h[1] = (__bf16)a.y; r.h[2] = (__bf16)a.z; r.h[3] = (__bf16)a.w;
    r.h[4] = (__bf16)b.x; r.h[5] = (__bf16)b.y; r.h[6] = (__bf16)b.z; r.h[7] = (__bf16)b.w;
    return r.u;
}

// A fragment from a bf16 row (ISA 16-bit striping): element j ->
// k = kc + (j&7) + ((j>>3)<<4) + hi*8  ==  two contiguous 16-byte runs.
template<typename PT>
__device__ __forceinline__ v16bf load_frag_bf(PT row, int kc, int hi) {
    union { v16bf v; uint4 q[2]; } r;
    r.q[0] = *(const uint4*)((const __bf16*)row + kc + hi * 8);
    r.q[1] = *(const uint4*)((const __bf16*)row + kc + 16 + hi * 8);
    return r.v;
}

// ---------------- weight swizzle: W[K,NSRC] f32 -> fragment-major bf16 ----------
// Layout: [chunk=K/32][ntile=NOUT/16][lane 0..31][j 0..15]; a wave's fragment is
// 1KB contiguous, a lane's is 32B contiguous.
template<int KDIM, int NOUT, int NSRC>
__global__ void swz_w_kernel(const float* __restrict__ s, __bf16* __restrict__ d) {
    constexpr int NT = NOUT / 16;
    constexpr int total = (KDIM / 32) * NT * 32 * 16;
    int i = blockIdx.x * blockDim.x + threadIdx.x;
    if (i < total) {
        int j     = i & 15;
        int lane  = (i >> 4) & 31;
        int ntile = (i >> 9) % NT;
        int chunk = i / (512 * NT);
        int n  = ntile * 16 + (lane & 15);
        int hi = lane >> 4;
        int k  = chunk * 32 + (j & 7) + ((j >> 3) << 4) + hi * 8;
        d[i] = (n < NSRC) ? (__bf16)s[k * NSRC + n] : (__bf16)0.0f;
    }
}

// ---------------- degree ----------------
__global__ void deg_kernel(const int* __restrict__ dst, float* __restrict__ deg) {
    int e = blockIdx.x * blockDim.x + threadIdx.x;
    if (e < NEDGES) atomic_add_f32(&deg[dst[e]], 1.0f);
}

// acc[n,c] += deg[n] * bias[c]  (so acc/deg = mean_msg + bias when deg>0)
__global__ void fold_bias_kernel(const float* __restrict__ bias,
                                 const float* __restrict__ deg,
                                 float* __restrict__ acc) {
    long long i = (long long)blockIdx.x * blockDim.x + threadIdx.x;
    if (i < (long long)NNODES * 128) {
        int n = (int)(i >> 7), c = (int)(i & 127);
        acc[i] += deg[n] * bias[c];
    }
}

// ---------------- per-edge message GEMM + scatter accumulate ----------------
// A = [srcfeat[src[e]] || efeats[e]] (16 edges x KDIM), staged in LDS as bf16.
// ST = float (layer 1: nfeats) or __bf16 (layer 2: h1).
// block = 256 (8 waves); wave w owns output columns [16w,16w+16).
template<int KDIM, int D1, typename ST>
__global__ __launch_bounds__(256)
void msg_gemm_scatter(const ST* __restrict__ srcfeat,     // [N, D1]
                      const float* __restrict__ ef,       // [E, 64] f32
                      const int*   __restrict__ src,
                      const int*   __restrict__ dst,
                      const __bf16* __restrict__ Wswz,    // fragment-major
                      float* __restrict__ acc)            // [N, 128] f32
{
    __shared__ __bf16 As[16][KDIM + 8];                   // stride = 16 mod 128B
    const long long e0 = (long long)blockIdx.x * 16;

    for (int idx = threadIdx.x; idx < 16 * (KDIM / 8); idx += 256) {
        int row = idx / (KDIM / 8), g = idx % (KDIM / 8);
        int k = g * 8;
        long long e = e0 + row;
        uint4 out;
        if (k < D1) {
            if constexpr (sizeof(ST) == 2) {
                out = *(const uint4*)((const __bf16*)srcfeat + (long long)src[e] * D1 + k);
            } else {
                const float* srow = (const float*)srcfeat + (long long)src[e] * D1 + k;
                out = cvt8(*(const float4*)srow, *(const float4*)(srow + 4));
            }
        } else {
            const float* erow = ef + e * EDIMC + (k - D1);
            out = cvt8(*(const float4*)erow, *(const float4*)(erow + 4));
        }
        *(uint4*)&As[row][k] = out;
    }
    __syncthreads();

    const int lane = threadIdx.x & 31;
    const int wave = threadIdx.x >> 5;
    const int m  = lane & 15;
    const int hi = lane >> 4;
    const int ncol = wave * 16 + (lane & 15);

    v8f c = {};
    #pragma unroll
    for (int kc = 0; kc < KDIM; kc += 32) {
        v16bf a = load_frag_bf(&As[m][0], kc, hi);
        v16bf b = *(const v16bf*)(Wswz + ((size_t)((kc >> 5) * 8 + wave) << 9) + (lane << 4));
        c = __builtin_amdgcn_wmma_f32_16x16x32_bf16(false, a, false, b,
                                                    (short)0, c, false, false);
    }

    #pragma unroll
    for (int r = 0; r < 8; ++r) {
        long long e = e0 + r + hi * 8;          // C/D row = r + hi*8
        int d = dst[e];
        atomic_add_f32(&acc[(long long)d * DOUTC + ncol], c[r]);
    }
}

// ---------------- node apply GEMM: h_out = relu([h || acc/deg] @ Wa + b) --------
// ST = float (layer 1: nfeats) or __bf16 (layer 2: h1); output stored bf16.
template<int KDIM, int DH, typename ST>
__global__ __launch_bounds__(256)
void apply_gemm(const ST* __restrict__ hprev,      // [N, DH]
                const float* __restrict__ acc,     // [N, 128] (bias pre-folded)
                const float* __restrict__ deg,     // [N]
                const __bf16* __restrict__ Wswz,   // fragment-major
                const float* __restrict__ bias,    // [128]
                __bf16* __restrict__ hout)         // [N, 128] bf16
{
    __shared__ __bf16 As[16][KDIM + 8];
    const long long n0 = (long long)blockIdx.x * 16;

    for (int idx = threadIdx.x; idx < 16 * (KDIM / 8); idx += 256) {
        int row = idx / (KDIM / 8), g = idx % (KDIM / 8);
        int k = g * 8;
        long long node = n0 + row;
        uint4 out;
        if (k < DH) {
            if constexpr (sizeof(ST) == 2) {
                out = *(const uint4*)((const __bf16*)hprev + node * DH + k);
            } else {
                const float* srow = (const float*)hprev + node * DH + k;
                out = cvt8(*(const float4*)srow, *(const float4*)(srow + 4));
            }
        } else {
            float inv = 1.0f / fmaxf(deg[node], 1.0f);
            const float* arow = acc + node * DOUTC + (k - DH);
            float4 f0 = *(const float4*)arow;
            float4 f1 = *(const float4*)(arow + 4);
            f0.x *= inv; f0.y *= inv; f0.z *= inv; f0.w *= inv;
            f1.x *= inv; f1.y *= inv; f1.z *= inv; f1.w *= inv;
            out = cvt8(f0, f1);
        }
        *(uint4*)&As[row][k] = out;
    }
    __syncthreads();

    const int lane = threadIdx.x & 31;
    const int wave = threadIdx.x >> 5;
    const int m  = lane & 15;
    const int hi = lane >> 4;
    const int ncol = wave * 16 + (lane & 15);

    v8f c = {};
    #pragma unroll
    for (int kc = 0; kc < KDIM; kc += 32) {
        v16bf a = load_frag_bf(&As[m][0], kc, hi);
        v16bf b = *(const v16bf*)(Wswz + ((size_t)((kc >> 5) * 8 + wave) << 9) + (lane << 4));
        c = __builtin_amdgcn_wmma_f32_16x16x32_bf16(false, a, false, b,
                                                    (short)0, c, false, false);
    }

    float bv = bias[ncol];
    #pragma unroll
    for (int r = 0; r < 8; ++r) {
        long long node = n0 + r + hi * 8;
        hout[node * DOUTC + ncol] = (__bf16)fmaxf(c[r] + bv, 0.0f);
    }
}

// ---------------- edge predictor: score = [h2[src] || h2[dst]] @ Wp + bp --------
// block = 256 = 8 waves; each wave owns a 16-edge tile; K=256, one 16-wide N tile.
// A fragments gathered directly from bf16 h2 rows (2x b128 per chunk per lane).
__global__ __launch_bounds__(256)
void predictor_kernel(const __bf16* __restrict__ h2,      // [N, 128] bf16
                      const int* __restrict__ src,
                      const int* __restrict__ dst,
                      const __bf16* __restrict__ Wswz,    // [8][32][16] fragment-major
                      const float* __restrict__ bp,       // [10]
                      float* __restrict__ out)            // [E, 10] f32
{
    const int lane = threadIdx.x & 31;
    const int wave = threadIdx.x >> 5;
    const long long e0 = ((long long)blockIdx.x * 8 + wave) * 16;
    const int m  = lane & 15;
    const int hi = lane >> 4;

    const long long em = e0 + m;
    const __bf16* rowS = h2 + (long long)src[em] * DOUTC;
    const __bf16* rowD = h2 + (long long)dst[em] * DOUTC;

    v8f c = {};
    #pragma unroll
    for (int kc = 0; kc < 2 * DOUTC; kc += 32) {
        // chunks 0..3 read the src row (k<128), chunks 4..7 the dst row
        const __bf16* row = (kc < DOUTC) ? rowS : (rowD - DOUTC);
        v16bf a = load_frag_bf(row, kc, hi);
        v16bf b = *(const v16bf*)(Wswz + ((size_t)(kc >> 5) << 9) + (lane << 4));
        c = __builtin_amdgcn_wmma_f32_16x16x32_bf16(false, a, false, b,
                                                    (short)0, c, false, false);
    }

    const int ncol = lane & 15;
    if (ncol < NCLS) {
        float bv = bp[ncol];
        #pragma unroll
        for (int r = 0; r < 8; ++r) {
            long long e = e0 + r + hi * 8;
            out[e * NCLS + ncol] = c[r] + bv;
        }
    }
}

// ---------------- host launch ----------------
extern "C" void kernel_launch(void* const* d_in, const int* in_sizes, int n_in,
                              void* d_out, int out_size, void* d_ws, size_t ws_size,
                              hipStream_t stream) {
    const float* nfeats = (const float*)d_in[0];
    const float* efeats = (const float*)d_in[1];
    const int*   src    = (const int*)  d_in[2];
    const int*   dst    = (const int*)  d_in[3];
    const float* Wm1    = (const float*)d_in[4];
    const float* bm1    = (const float*)d_in[5];
    const float* Wa1    = (const float*)d_in[6];
    const float* ba1    = (const float*)d_in[7];
    const float* Wm2    = (const float*)d_in[8];
    const float* bm2    = (const float*)d_in[9];
    const float* Wa2    = (const float*)d_in[10];
    const float* ba2    = (const float*)d_in[11];
    const float* Wp     = (const float*)d_in[12];
    const float* bp     = (const float*)d_in[13];
    float* score = (float*)d_out;

    char* ws = (char*)d_ws;
    __bf16* Wm1b = (__bf16*)ws; ws += 128 * 128 * 2;
    __bf16* Wa1b = (__bf16*)ws; ws += 192 * 128 * 2;
    __bf16* Wm2b = (__bf16*)ws; ws += 192 * 128 * 2;
    __bf16* Wa2b = (__bf16*)ws; ws += 256 * 128 * 2;
    __bf16* Wpb  = (__bf16*)ws; ws += 256 * 16  * 2;
    float*  deg  = (float*)ws;  ws += ((NNODES * 4 + 255) / 256) * 256;
    float*  acc  = (float*)ws;  ws += (size_t)NNODES * 128 * 4;
    __bf16* h1   = (__bf16*)ws; ws += (size_t)NNODES * 128 * 2;
    __bf16* h2   = (__bf16*)ws; ws += (size_t)NNODES * 128 * 2;

    // one-time weight swizzle into WMMA fragment layout (tiny)
    swz_w_kernel<128, 128, 128><<<(128 * 128 + 255) / 256, 256, 0, stream>>>(Wm1, Wm1b);
    swz_w_kernel<192, 128, 128><<<(192 * 128 + 255) / 256, 256, 0, stream>>>(Wa1, Wa1b);
    swz_w_kernel<192, 128, 128><<<(192 * 128 + 255) / 256, 256, 0, stream>>>(Wm2, Wm2b);
    swz_w_kernel<256, 128, 128><<<(256 * 128 + 255) / 256, 256, 0, stream>>>(Wa2, Wa2b);
    swz_w_kernel<256, 16,  10><<<(256 * 16 + 255) / 256, 256, 0, stream>>>(Wp, Wpb);

    hipMemsetAsync(deg, 0, NNODES * sizeof(float), stream);
    hipMemsetAsync(acc, 0, (size_t)NNODES * 128 * sizeof(float), stream);

    deg_kernel<<<(NEDGES + 255) / 256, 256, 0, stream>>>(dst, deg);

    // ---- layer 1 ----
    msg_gemm_scatter<128, 64, float><<<NEDGES / 16, 256, 0, stream>>>(
        nfeats, efeats, src, dst, Wm1b, acc);
    fold_bias_kernel<<<(NNODES * 128 + 255) / 256, 256, 0, stream>>>(bm1, deg, acc);
    apply_gemm<192, 64, float><<<NNODES / 16, 256, 0, stream>>>(
        nfeats, acc, deg, Wa1b, ba1, h1);

    // ---- layer 2 ----
    hipMemsetAsync(acc, 0, (size_t)NNODES * 128 * sizeof(float), stream);
    msg_gemm_scatter<192, 128, __bf16><<<NEDGES / 16, 256, 0, stream>>>(
        h1, efeats, src, dst, Wm2b, acc);
    fold_bias_kernel<<<(NNODES * 128 + 255) / 256, 256, 0, stream>>>(bm2, deg, acc);
    apply_gemm<256, 128, __bf16><<<NNODES / 16, 256, 0, stream>>>(
        h1, acc, deg, Wa2b, ba2, h2);

    // ---- predictor ----
    predictor_kernel<<<NEDGES / 128, 256, 0, stream>>>(h2, src, dst, Wpb, bp, score);

    (void)in_sizes; (void)n_in; (void)out_size; (void)ws_size;
}
